// LSTMclassifier_266287972947
// MI455X (gfx1250) — compile-verified, matrix operand order checked by reading, not run on hardware
//
#include <hip/hip_runtime.h>
#include <hip/hip_bf16.h>
#include <math.h>

// ---------------------------------------------------------------------------
// LSTM classifier on gfx1250 (MI455X), bf16 WMMA, fp32 accumulate.
// B=256, T=512, EMB=300, HID=300, N_GRAM=2 -> F=600, gates=1200, NCLS=5.
// Padded dims: HIDP=304 (19x16), NGP=1216 (=4*304, 76 n-tiles),
//              K_ih: 600->608 (19x32), K_hh: 300->320 (10x32).
// ---------------------------------------------------------------------------

typedef __attribute__((ext_vector_type(16))) __bf16 v16bf;
typedef __attribute__((ext_vector_type(8)))  __bf16 v8bf;
typedef __attribute__((ext_vector_type(8)))  float  v8f;

#define B_      256
#define T_      512
#define EMB_    300
#define HID_    300
#define F_      608      // padded 2*EMB
#define KT_IH   19       // 608/32
#define KT_HH   10       // 320/32
#define HIDP_   304
#define NGP_    1216     // 4*HIDP
#define NT_     76       // NGP/16
#define JT_     19       // HIDP/16
#define HSTRIDE 320      // padded h row (bf16 elements)

__device__ __forceinline__ unsigned short f2bf(float f) {
    unsigned int u = __float_as_uint(f);
    u += 0x7FFFu + ((u >> 16) & 1u);          // round-to-nearest-even
    return (unsigned short)(u >> 16);
}
__device__ __forceinline__ float sigmoidf_(float x) {
    return 1.0f / (1.0f + __expf(-x));
}
// K index inside a 32-deep fragment for (lane, element e) — 16-bit A/B layout:
// lanes 0-15 hold K {0..7,16..23}, lanes 16-31 hold K {8..15,24..31}.
__device__ __forceinline__ int frag_k(int lane, int e) {
    return ((e >> 3) << 4) + ((lane >> 4) << 3) + (e & 7);
}
// Build a v16bf A-fragment from a khalf-adjusted row pointer: two 16B chunks
// at immediate offsets (c0 = kt*32 elements -> kt*64 bytes).
__device__ __forceinline__ v16bf load_afrag(const __bf16* rowp, int c0) {
    v8bf lo = *(const v8bf*)(rowp + c0);
    v8bf hi = *(const v8bf*)(rowp + c0 + 16);
    v16bf r;
#pragma unroll
    for (int i = 0; i < 8; ++i) { r[i] = lo[i]; r[i + 8] = hi[i]; }
    return r;
}

// --------------------------- weight packing --------------------------------
// W_ih [1200][600] -> WihP [NT_][KT_IH] blocks of 32x16 bf16 in B-fragment
// lane order (lane*16 + e). n-tile = gate*19 + jtile, zero-padded.
__global__ void pack_wih_kernel(const float* __restrict__ W, unsigned short* __restrict__ P) {
    int blk = blockIdx.x;                 // ntile*KT_IH + kt
    int ntile = blk / KT_IH, kt = blk % KT_IH;
    int g = ntile / JT_, j = ntile % JT_;
    int tid = threadIdx.x;                // 0..511
    int lane = tid >> 4, e = tid & 15;
    int nl = lane & 15;
    int kl = frag_k(lane, e);
    int hid = j * 16 + nl;
    int kcol = kt * 32 + kl;
    float v = 0.0f;
    if (hid < HID_ && kcol < 600) v = W[(size_t)(g * HID_ + hid) * 600 + kcol];
    P[(size_t)blk * 512 + lane * 16 + e] = f2bf(v);
}

// W_hh [1200][300] -> WhhP [NT_][KT_HH] blocks.
__global__ void pack_whh_kernel(const float* __restrict__ W, unsigned short* __restrict__ P) {
    int blk = blockIdx.x;                 // ntile*KT_HH + kt
    int ntile = blk / KT_HH, kt = blk % KT_HH;
    int g = ntile / JT_, j = ntile % JT_;
    int tid = threadIdx.x;
    int lane = tid >> 4, e = tid & 15;
    int nl = lane & 15;
    int kl = frag_k(lane, e);
    int hid = j * 16 + nl;
    int kcol = kt * 32 + kl;
    float v = 0.0f;
    if (hid < HID_ && kcol < HID_) v = W[(size_t)(g * HID_ + hid) * HID_ + kcol];
    P[(size_t)blk * 512 + lane * 16 + e] = f2bf(v);
}

// biasP[NGP_]: gate-padded b_ih + b_hh.
__global__ void pack_bias_kernel(const float* __restrict__ bih, const float* __restrict__ bhh,
                                 float* __restrict__ biasP) {
    int n = blockIdx.x * blockDim.x + threadIdx.x;
    if (n >= NGP_) return;
    int ntile = n >> 4, nl = n & 15;
    int g = ntile / JT_, j = ntile % JT_;
    int hid = j * 16 + nl;
    biasP[n] = (hid < HID_) ? (bih[g * HID_ + hid] + bhh[g * HID_ + hid]) : 0.0f;
}

// Zero h double-buffer (incl. K padding), c state, h accumulator.
__global__ void init_state_kernel(unsigned short* __restrict__ H0, unsigned short* __restrict__ H1,
                                  float* __restrict__ C, float* __restrict__ S) {
    int i = blockIdx.x * blockDim.x + threadIdx.x;
    if (i < B_ * HSTRIDE) { H0[i] = 0; H1[i] = 0; }
    if (i < B_ * HIDP_)   { C[i] = 0.0f; S[i] = 0.0f; }
}

// ------------------------- embedding gather --------------------------------
// X[(t*256+b)][F_] = bf16 concat(emb'(b,t), emb'(b,t-1)), emb'(b,T-1)=0,
// emb'(b,-1)=0, columns 600..607 zero.
__global__ void gather_x_kernel(const int* __restrict__ ids, const float* __restrict__ glove,
                                unsigned short* __restrict__ X) {
    int row = blockIdx.x;                 // t*256 + b
    int t = row >> 8, b = row & 255;
    unsigned short* xr = X + (size_t)row * F_;
    int id0 = ids[b * T_ + t];
    int id1 = (t > 0) ? ids[b * T_ + t - 1] : 0;
    bool z0 = (t == T_ - 1);
    bool z1 = (t == 0);
    for (int k = threadIdx.x; k < F_; k += blockDim.x) {
        float v = 0.0f;
        if (k < EMB_)            { if (!z0) v = glove[(size_t)id0 * EMB_ + k]; }
        else if (k < 2 * EMB_)   { if (!z1) v = glove[(size_t)id1 * EMB_ + (k - EMB_)]; }
        xr[k] = f2bf(v);
    }
}

// ----------------------- big input-projection GEMM -------------------------
// G[131072][NGP_] = X @ W_ih^T + bias  (bf16 x bf16 -> fp32 via WMMA)
// Each wave computes TWO M-tiles (32 rows) sharing one B fragment.
__global__ void input_gemm_kernel(const unsigned short* __restrict__ X_,
                                  const unsigned short* __restrict__ Wp_,
                                  const float* __restrict__ biasP,
                                  float* __restrict__ G) {
    const __bf16* X  = (const __bf16*)X_;
    const __bf16* Wp = (const __bf16*)Wp_;
    int lane  = threadIdx.x & 31;
    int w     = threadIdx.x >> 5;
    int ntile = blockIdx.y * 4 + w;       // 0..75 (76 = 19*4 exact)
    int mbase = blockIdx.x * 32;          // two 16-row tiles per wave
    int n     = lane & 15;                 // C column / A row within tile
    int khalf = lane >> 4;
    // khalf folded into the row pointer -> all A offsets are immediates.
    const __bf16* ap0 = X + (size_t)(mbase + n) * F_ + khalf * 8;
    const __bf16* ap1 = ap0 + (size_t)16 * F_;
    // hoisted weight base -> kt*512 halves = kt*1024 B immediate offsets.
    const __bf16* wbase = Wp + (size_t)ntile * (KT_IH * 512) + lane * 16;
    float bv = biasP[ntile * 16 + n];
    v8f acc0, acc1;
#pragma unroll
    for (int r = 0; r < 8; ++r) { acc0[r] = bv; acc1[r] = bv; }
#pragma unroll
    for (int kt = 0; kt < KT_IH; ++kt) {
        v16bf bf  = *(const v16bf*)(wbase + kt * 512);
        v16bf af0 = load_afrag(ap0, kt * 32);
        v16bf af1 = load_afrag(ap1, kt * 32);
        acc0 = __builtin_amdgcn_wmma_f32_16x16x32_bf16(false, af0, false, bf,
                                                       (short)0, acc0, false, false);
        acc1 = __builtin_amdgcn_wmma_f32_16x16x32_bf16(false, af1, false, bf,
                                                       (short)0, acc1, false, false);
    }
    float* grow = G + (size_t)mbase * NGP_ + ntile * 16;
    int mrow = 8 * khalf;
#pragma unroll
    for (int r = 0; r < 8; ++r) {
        grow[(size_t)(r + mrow) * NGP_ + n]      = acc0[r];
        grow[(size_t)(r + mrow + 16) * NGP_ + n] = acc1[r];
    }
}

// --------------------------- fused LSTM step -------------------------------
// Block = (mtile, jtile): 4 waves, wave g computes gate-g tile via WMMA with
// C preloaded from G[t]; LDS exchange; 128 threads do the cell update.
__global__ void lstm_step_kernel(const float* __restrict__ G,
                                 const unsigned short* __restrict__ Wp_,
                                 const unsigned short* __restrict__ Hr_,
                                 unsigned short* __restrict__ Hw,
                                 float* __restrict__ Cst,
                                 float* __restrict__ hsum,
                                 int t) {
    const __bf16* Hr = (const __bf16*)Hr_;
    const __bf16* Wp = (const __bf16*)Wp_;
    __shared__ float gt[4][16][16];
    int tid   = threadIdx.x;
    int lane  = tid & 31;
    int g     = tid >> 5;                 // gate 0..3 (i,f,g,o)
    int mtile = blockIdx.x;               // 0..15
    int jtile = blockIdx.y;               // 0..18
    int n     = lane & 15;
    int khalf = lane >> 4;
    int ntile = g * JT_ + jtile;

    const float* grow = G + ((size_t)t * B_ + mtile * 16) * NGP_ + ntile * 16;
    v8f acc;
#pragma unroll
    for (int r = 0; r < 8; ++r) acc[r] = grow[(size_t)(r + 8 * khalf) * NGP_ + n];

    const __bf16* ap = Hr + (size_t)(mtile * 16 + n) * HSTRIDE + khalf * 8;
    const __bf16* wbase = Wp + (size_t)ntile * (KT_HH * 512) + lane * 16;
#pragma unroll
    for (int kt = 0; kt < KT_HH; ++kt) {
        v16bf af = load_afrag(ap, kt * 32);
        v16bf bf = *(const v16bf*)(wbase + kt * 512);
        acc = __builtin_amdgcn_wmma_f32_16x16x32_bf16(false, af, false, bf,
                                                      (short)0, acc, false, false);
    }
#pragma unroll
    for (int r = 0; r < 8; ++r) gt[g][r + 8 * khalf][n] = acc[r];
    __syncthreads();

#pragma unroll
    for (int ee = 0; ee < 2; ++ee) {
        int e = tid + ee * 128;           // 0..255 element of the 16x16 tile
        int m = e >> 4, c = e & 15;
        int b = mtile * 16 + m;
        int hid = jtile * 16 + c;
        if (hid < HID_) {
            float iv = sigmoidf_(gt[0][m][c]);
            float fv = sigmoidf_(gt[1][m][c]);
            float gv = tanhf(gt[2][m][c]);
            float ov = sigmoidf_(gt[3][m][c]);
            float cc = fv * Cst[b * HIDP_ + hid] + iv * gv;
            Cst[b * HIDP_ + hid] = cc;
            float hh = ov * tanhf(cc);
            hsum[b * HIDP_ + hid] += hh;
            Hw[b * HSTRIDE + hid] = f2bf(hh);
        }
    }
}

// ------------------------------ classifier ---------------------------------
__global__ void classifier_kernel(const float* __restrict__ hsum,
                                  const float* __restrict__ W1, const float* __restrict__ b1,
                                  const float* __restrict__ W2, const float* __restrict__ b2,
                                  float* __restrict__ out) {
    int b = blockIdx.x;
    __shared__ float l1[100];
    __shared__ float l2[5];
    int tid = threadIdx.x;
    if (tid < 100) {
        float s = b1[tid];
        const float* w = W1 + tid * HID_;
        const float* p = hsum + b * HIDP_;
        const float inv_t = 1.0f / (float)T_;
        for (int k = 0; k < HID_; ++k) s += w[k] * (p[k] * inv_t);
        l1[tid] = sigmoidf_(s);
    }
    __syncthreads();
    if (tid < 5) {
        float s = b2[tid];
        const float* w = W2 + tid * 100;
        for (int k = 0; k < 100; ++k) s += w[k] * l1[k];
        l2[tid] = sigmoidf_(s);
    }
    __syncthreads();
    if (tid < 5) {
        float mx = l2[0];
        for (int k = 1; k < 5; ++k) mx = fmaxf(mx, l2[k]);
        float se = 0.0f;
        for (int k = 0; k < 5; ++k) se += expf(l2[k] - mx);
        out[b * 5 + tid] = l2[tid] - mx - logf(se);
    }
}

// ------------------------------ workspace map ------------------------------
constexpr size_t aln(size_t x) { return (x + 255) & ~(size_t)255; }
constexpr size_t SZ_X    = (size_t)B_ * T_ * F_ * 2;        // 159.4 MB bf16
constexpr size_t SZ_G    = (size_t)B_ * T_ * NGP_ * 4;      // 637.5 MB fp32
constexpr size_t SZ_WIHP = (size_t)NT_ * KT_IH * 512 * 2;
constexpr size_t SZ_WHHP = (size_t)NT_ * KT_HH * 512 * 2;
constexpr size_t SZ_BIAS = (size_t)NGP_ * 4;
constexpr size_t SZ_H    = (size_t)B_ * HSTRIDE * 2;
constexpr size_t SZ_CS   = (size_t)B_ * HIDP_ * 4;

constexpr size_t OFF_X    = 0;
constexpr size_t OFF_G    = aln(OFF_X + SZ_X);
constexpr size_t OFF_WIHP = aln(OFF_G + SZ_G);
constexpr size_t OFF_WHHP = aln(OFF_WIHP + SZ_WIHP);
constexpr size_t OFF_BIAS = aln(OFF_WHHP + SZ_WHHP);
constexpr size_t OFF_H0   = aln(OFF_BIAS + SZ_BIAS);
constexpr size_t OFF_H1   = aln(OFF_H0 + SZ_H);
constexpr size_t OFF_C    = aln(OFF_H1 + SZ_H);
constexpr size_t OFF_HS   = aln(OFF_C + SZ_CS);

extern "C" void kernel_launch(void* const* d_in, const int* in_sizes, int n_in,
                              void* d_out, int out_size, void* d_ws, size_t ws_size,
                              hipStream_t stream) {
    (void)in_sizes; (void)n_in; (void)out_size; (void)ws_size;
    const int*   ids   = (const int*)d_in[0];
    // d_in[1] = max_num_of_words (compile-time T_=512)
    const float* glove = (const float*)d_in[2];
    const float* Wih   = (const float*)d_in[3];
    const float* Whh   = (const float*)d_in[4];
    const float* bih   = (const float*)d_in[5];
    const float* bhh   = (const float*)d_in[6];
    const float* W1    = (const float*)d_in[7];
    const float* b1    = (const float*)d_in[8];
    const float* W2    = (const float*)d_in[9];
    const float* b2    = (const float*)d_in[10];
    float* out = (float*)d_out;
    char*  ws  = (char*)d_ws;

    unsigned short* X     = (unsigned short*)(ws + OFF_X);
    float*          G     = (float*)(ws + OFF_G);
    unsigned short* WihP  = (unsigned short*)(ws + OFF_WIHP);
    unsigned short* WhhP  = (unsigned short*)(ws + OFF_WHHP);
    float*          biasP = (float*)(ws + OFF_BIAS);
    unsigned short* H0    = (unsigned short*)(ws + OFF_H0);
    unsigned short* H1    = (unsigned short*)(ws + OFF_H1);
    float*          Cst   = (float*)(ws + OFF_C);
    float*          hsum  = (float*)(ws + OFF_HS);

    // 1. pack weights + bias, zero state
    pack_wih_kernel<<<NT_ * KT_IH, 512, 0, stream>>>(Wih, WihP);
    pack_whh_kernel<<<NT_ * KT_HH, 512, 0, stream>>>(Whh, WhhP);
    pack_bias_kernel<<<(NGP_ + 255) / 256, 256, 0, stream>>>(bih, bhh, biasP);
    init_state_kernel<<<(B_ * HSTRIDE + 255) / 256, 256, 0, stream>>>(H0, H1, Cst, hsum);

    // 2. embedding gather into padded bf16 feature matrix
    gather_x_kernel<<<B_ * T_, 256, 0, stream>>>(ids, glove, X);

    // 3. one big WMMA GEMM: all input projections + bias, out of the time loop
    dim3 ggrid((B_ * T_) / 32, NT_ / 4);
    input_gemm_kernel<<<ggrid, 128, 0, stream>>>(X, WihP, biasP, G);

    // 4. 512 sequential fused LSTM steps (double-buffered h)
    dim3 sgrid(B_ / 16, JT_);
    for (int t = 0; t < T_; ++t) {
        const unsigned short* hr = (t & 1) ? H1 : H0;
        unsigned short*       hw = (t & 1) ? H0 : H1;
        lstm_step_kernel<<<sgrid, 128, 0, stream>>>(G, WhhP, hr, hw, Cst, hsum, t);
    }

    // 5. mean-pool + MLP head + log_softmax
    classifier_kernel<<<B_, 128, 0, stream>>>(hsum, W1, b1, W2, b2, out);
}
